// Geo3DGNN_44306882625631
// MI455X (gfx1250) — matrix operand order repaired and checked by reading, not compiled, not run on hardware
//
#include <hip/hip_runtime.h>

typedef __attribute__((ext_vector_type(16))) _Float16 v16h;
typedef __attribute__((ext_vector_type(4)))  _Float16 v4h;
typedef __attribute__((ext_vector_type(8)))  float    v8f;

#define H_DIM    128
#define SP_DIM   32
#define FE_IN    16
#define FE_OUT   64
#define FUSED_DIM 96
#define OUT_DIM  10
#define EPSV     1e-5f

__device__ __forceinline__ void atomAddF(float* p, float v) {
  // lowers to global_atomic_add_f32 (no-return) — lands in L2; accumulators fit in 192MB L2
  __hip_atomic_fetch_add(p, v, __ATOMIC_RELAXED, __HIP_MEMORY_SCOPE_AGENT);
}

// ---------------------------------------------------------------- utility

__global__ void fill_kernel(float* __restrict__ p, float v, long n) {
  long i = (long)blockIdx.x * blockDim.x + threadIdx.x;
  if (i < n) p[i] = v;
}

// Pack f32 row-major W[K x ncols] into WMMA B-fragment order (f16):
// fragment (nt, kt) for wave-column-tile nt, k-tile kt; within it: lane ln = (n%16) + 16*(((k%32)>>3)&1),
// element e = 8*((k%32)>>4) + k%8.  Stored as v16h blocks -> one contiguous 32B load per lane in the GEMM.
__global__ void pack_b_kernel(const float* __restrict__ W, _Float16* __restrict__ out,
                              int K, int ncols) {
  int idx = blockIdx.x * blockDim.x + threadIdx.x;
  if (idx >= K * ncols) return;
  int k  = idx / ncols;
  int nn = idx - k * ncols;
  int nt = nn >> 4;
  int kt = k >> 5, kk = k & 31;
  int ln = (nn & 15) + (((kk >> 3) & 1) << 4);
  int e  = ((kk & 16) ? 8 : 0) + (kk & 7);
  int KT = K >> 5;
  out[(size_t)(((nt * KT) + kt) * 32 + ln) * 16 + e] = (_Float16)W[idx];
}

// ---------------------------------------------------------------- node embedding (3->32->32 MLP, 16->64 MLP)

__global__ void embed_kernel(const float* __restrict__ x,
                             const float* __restrict__ sp_w1, const float* __restrict__ sp_b1,
                             const float* __restrict__ sp_w2, const float* __restrict__ sp_b2,
                             const float* __restrict__ fe_w,  const float* __restrict__ fe_b,
                             float* __restrict__ h0, int n) {
  __shared__ float W1[3*SP_DIM], B1[SP_DIM], W2[SP_DIM*SP_DIM], B2[SP_DIM];
  __shared__ float FW[FE_IN*FE_OUT], FB[FE_OUT];
  for (int i = threadIdx.x; i < 3*SP_DIM; i += blockDim.x) W1[i] = sp_w1[i];
  for (int i = threadIdx.x; i < SP_DIM; i += blockDim.x) { B1[i] = sp_b1[i]; B2[i] = sp_b2[i]; }
  for (int i = threadIdx.x; i < SP_DIM*SP_DIM; i += blockDim.x) W2[i] = sp_w2[i];
  for (int i = threadIdx.x; i < FE_IN*FE_OUT; i += blockDim.x) FW[i] = fe_w[i];
  for (int i = threadIdx.x; i < FE_OUT; i += blockDim.x) FB[i] = fe_b[i];
  __syncthreads();
  int node = blockIdx.x * blockDim.x + threadIdx.x;
  if (node >= n) return;
  const float* xr = x + (long)node * 19;
  float c0 = xr[0], c1 = xr[1], c2 = xr[2];
  float t1[SP_DIM];
  #pragma unroll
  for (int j = 0; j < SP_DIM; ++j) {
    float s = B1[j] + c0 * W1[0*SP_DIM + j] + c1 * W1[1*SP_DIM + j] + c2 * W1[2*SP_DIM + j];
    t1[j] = fmaxf(s, 0.f);
  }
  float* hr = h0 + (long)node * FUSED_DIM;
  for (int j = 0; j < SP_DIM; ++j) {
    float s = B2[j];
    #pragma unroll
    for (int k = 0; k < SP_DIM; ++k) s += t1[k] * W2[k*SP_DIM + j];
    hr[j] = fmaxf(s, 0.f);
  }
  float f[FE_IN];
  #pragma unroll
  for (int k = 0; k < FE_IN; ++k) f[k] = xr[3 + k];
  for (int j = 0; j < FE_OUT; ++j) {
    float s = FB[j];
    #pragma unroll
    for (int k = 0; k < FE_IN; ++k) s += f[k] * FW[k*FE_OUT + j];
    hr[SP_DIM + j] = fmaxf(s, 0.f);
  }
}

// ---------------------------------------------------------------- degrees / norms

__global__ void degree_kernel(const int* __restrict__ dst, const float* __restrict__ ew,
                              float* __restrict__ cnt, float* __restrict__ deg, int E) {
  int e = blockIdx.x * blockDim.x + threadIdx.x;
  if (e >= E) return;
  int d = dst[e];
  atomAddF(&cnt[d], 1.0f);
  atomAddF(&deg[d], ew[e]);
}

__global__ void node_prep_kernel(const float* __restrict__ cnt, const float* __restrict__ deg,
                                 float* __restrict__ invcnt, float* __restrict__ dis, int n) {
  int i = blockIdx.x * blockDim.x + threadIdx.x;
  if (i >= n) return;
  invcnt[i] = 1.f / fmaxf(cnt[i], 1.f);
  dis[i] = rsqrtf(deg[i]);   // deg >= 1 (self loop folded into init)
}

// ---------------------------------------------------------------- edge scatter (SAGE sum)
// one thread per edge: indices loaded once, source row read as a 512B/384B sequential burst,
// scatter-adds resolve in L2 (accumulator is 25.6MB << 192MB L2)

template <int D>
__global__ void scatter_sum_kernel(const float* __restrict__ h, const int* __restrict__ src,
                                   const int* __restrict__ dst, float* __restrict__ acc, int E) {
  int e = blockIdx.x * blockDim.x + threadIdx.x;
  if (e >= E) return;
  int s = src[e], d = dst[e];
  const float* hs = h + (long)s * D;
  float* p = acc + (long)d * D;
  #pragma unroll
  for (int g = 0; g < D / 4; ++g) {
    const float4 v = *(const float4*)(hs + g * 4);
    atomAddF(p + g*4 + 0, v.x); atomAddF(p + g*4 + 1, v.y);
    atomAddF(p + g*4 + 2, v.z); atomAddF(p + g*4 + 3, v.w);
  }
}

// ---------------------------------------------------------------- GCN: self-loop + bias init, then normalized scatter

__global__ void gcn_selfinit_kernel(const float* __restrict__ hw, const float* __restrict__ dis,
                                    const float* __restrict__ bias, float* __restrict__ acc, int n) {
  long idx = (long)blockIdx.x * blockDim.x + threadIdx.x;
  long total = (long)n * (H_DIM / 4);
  if (idx >= total) return;
  int i = (int)(idx >> 5);          // H_DIM/4 == 32
  int g = (int)(idx & 31);
  float d2 = dis[i] * dis[i];
  const float4 v = *(const float4*)(hw + (long)i * H_DIM + g * 4);
  float4 o;
  o.x = bias[g*4 + 0] + v.x * d2;
  o.y = bias[g*4 + 1] + v.y * d2;
  o.z = bias[g*4 + 2] + v.z * d2;
  o.w = bias[g*4 + 3] + v.w * d2;
  *(float4*)(acc + (long)i * H_DIM + g * 4) = o;
}

__global__ void gcn_scatter_kernel(const float* __restrict__ hw, const int* __restrict__ src,
                                   const int* __restrict__ dst, const float* __restrict__ ew,
                                   const float* __restrict__ dis, float* __restrict__ acc, int E) {
  int e = blockIdx.x * blockDim.x + threadIdx.x;
  if (e >= E) return;
  int s = src[e], d = dst[e];
  float nm = dis[s] * ew[e] * dis[d];
  const float* hs = hw + (long)s * H_DIM;
  float* p = acc + (long)d * H_DIM;
  #pragma unroll
  for (int g = 0; g < H_DIM / 4; ++g) {
    const float4 v = *(const float4*)(hs + g * 4);
    atomAddF(p + g*4 + 0, v.x * nm); atomAddF(p + g*4 + 1, v.y * nm);
    atomAddF(p + g*4 + 2, v.z * nm); atomAddF(p + g*4 + 3, v.w * nm);
  }
}

// ---------------------------------------------------------------- batchnorm

__global__ void bn_stats_kernel(const float* __restrict__ x, float* __restrict__ sum,
                                float* __restrict__ sq, int n) {
  int g = blockIdx.x * blockDim.x + threadIdx.x;
  int c = g & (H_DIM - 1);
  int r0 = g >> 7;
  int stride = (gridDim.x * blockDim.x) >> 7;
  float s = 0.f, q = 0.f;
  for (int r = r0; r < n; r += stride) {
    float v = x[(long)r * H_DIM + c];
    s += v; q += v * v;
  }
  atomAddF(&sum[c], s);
  atomAddF(&sq[c], q);
}

__global__ void bn_apply_relu_kernel(const float* __restrict__ x, const float* __restrict__ sum,
                                     const float* __restrict__ sq, const float* __restrict__ g,
                                     const float* __restrict__ b, float* __restrict__ y,
                                     long total, float invN) {
  long idx = (long)blockIdx.x * blockDim.x + threadIdx.x;
  if (idx >= total) return;
  int c = (int)(idx & (H_DIM - 1));
  float mu = sum[c] * invN;
  float var = sq[c] * invN - mu * mu;
  float v = (x[idx] - mu) * rsqrtf(var + EPSV) * g[c] + b[c];
  y[idx] = fmaxf(v, 0.f);
}

// ---------------------------------------------------------------- WMMA GEMM
// Y[n x ncols] = act( scale(A1)@W1 [+ A2@W2] + bias ), A f32 row-major [n x K],
// W pre-packed f16 fragments. Block = 16 rows, (ncols/16) waves, one 16x16 tile per wave.
// A tile is staged to LDS converted to f16 *in A-fragment order*; any aligned group of 4
// consecutive K elements lands in 4 contiguous fragment slots, so staging is
// global_load_b128 -> cvt -> ds_store_b64 per 4 elements. K-loop is then
// 1 aligned 32B LDS load + 1 contiguous 32B global load + 1 v_wmma per k-step.

template <int K, bool HAS2, bool RELU>
__global__ void wmma_gemm_kernel(const float* __restrict__ A1, const float* __restrict__ rscale,
                                 const v16h* __restrict__ W1p,
                                 const float* __restrict__ A2, const v16h* __restrict__ W2p,
                                 const float* __restrict__ bias,
                                 float* __restrict__ Y, int n, int ncols) {
  constexpr int KT = K / 32;
  constexpr int KV = K / 4;                // float4 chunks per row
  __shared__ v16h ldsA[KT * 32];           // one A fragment per (k-tile, lane): 32B each
  _Float16* lp = (_Float16*)ldsA;
  const int tid  = threadIdx.x;
  const int lane = tid & 31;
  const int wv   = tid >> 5;
  const int row0 = blockIdx.x * 16;
  const int nOut = wv * 16 + (lane & 15);

  v8f c = {};

  // ---- stage + swizzle-pack operand-1 A tile (f32 -> f16, fragment order, 4-wide)
  for (int i = tid; i < 16 * KV; i += blockDim.x) {
    int r = i / KV, kq = i - r * KV;       // K is a compile-time constant
    int k = kq * 4;
    int row = row0 + r;
    int rc = (row < n) ? row : (n - 1);    // clamp: N%16==0 in practice, keeps loads in-bounds
    float4 v = *(const float4*)(A1 + (long)rc * K + k);
    if (rscale) {
      float sc = rscale[rc];
      v.x *= sc; v.y *= sc; v.z *= sc; v.w *= sc;
    }
    int kt = k >> 5, kk = k & 31;
    int ln = r + (((kk >> 3) & 1) << 4);   // half-wave holding this K offset
    int e0 = 2 * (((kk & 16) ? 4 : 0) + ((kk & 7) >> 1));
    v4h hv;
    hv[0] = (_Float16)v.x; hv[1] = (_Float16)v.y;
    hv[2] = (_Float16)v.z; hv[3] = (_Float16)v.w;
    *(v4h*)&lp[(kt * 32 + ln) * 16 + e0] = hv;
  }
  if constexpr (HAS2) {
    // warm L2/WGP$ for operand-2 rows while operand 1 computes (global_prefetch_b8)
    __builtin_prefetch(&A2[(long)(row0 + (tid & 15)) * K], 0, 1);
  }
  __syncthreads();
  #pragma unroll
  for (int kt = 0; kt < KT; ++kt) {
    v16h a = ldsA[kt * 32 + lane];
    v16h b = W1p[(size_t)(wv * KT + kt) * 32 + lane];
    c = __builtin_amdgcn_wmma_f32_16x16x32_f16(false, a, false, b, (short)0, c, false, false);
  }

  if constexpr (HAS2) {
    __syncthreads();
    for (int i = tid; i < 16 * KV; i += blockDim.x) {
      int r = i / KV, kq = i - r * KV;
      int k = kq * 4;
      int row = row0 + r;
      int rc = (row < n) ? row : (n - 1);
      float4 v = *(const float4*)(A2 + (long)rc * K + k);
      int kt = k >> 5, kk = k & 31;
      int ln = r + (((kk >> 3) & 1) << 4);
      int e0 = 2 * (((kk & 16) ? 4 : 0) + ((kk & 7) >> 1));
      v4h hv;
      hv[0] = (_Float16)v.x; hv[1] = (_Float16)v.y;
      hv[2] = (_Float16)v.z; hv[3] = (_Float16)v.w;
      *(v4h*)&lp[(kt * 32 + ln) * 16 + e0] = hv;
    }
    __syncthreads();
    #pragma unroll
    for (int kt = 0; kt < KT; ++kt) {
      v16h a = ldsA[kt * 32 + lane];
      v16h b = W2p[(size_t)(wv * KT + kt) * 32 + lane];
      c = __builtin_amdgcn_wmma_f32_16x16x32_f16(false, a, false, b, (short)0, c, false, false);
    }
  }

  // ---- epilogue: C layout — VGPR r, lanes 0-15 -> M=r, lanes 16-31 -> M=r+8
  float bv = bias ? bias[nOut] : 0.f;
  const int mb = (lane >> 4) * 8;
  #pragma unroll
  for (int r = 0; r < 8; ++r) {
    int row = row0 + mb + r;
    if (row < n) {
      float v = c[r] + bv;
      if (RELU) v = fmaxf(v, 0.f);
      Y[(long)row * ncols + nOut] = v;
    }
  }
}

// ---------------------------------------------------------------- final 64 -> 10 classifier

__global__ void cls2_kernel(const float* __restrict__ z, const float* __restrict__ w,
                            const float* __restrict__ b, float* __restrict__ out, int n) {
  __shared__ float W[64 * OUT_DIM], B[OUT_DIM];
  for (int i = threadIdx.x; i < 64 * OUT_DIM; i += blockDim.x) W[i] = w[i];
  for (int i = threadIdx.x; i < OUT_DIM; i += blockDim.x) B[i] = b[i];
  __syncthreads();
  int node = blockIdx.x * blockDim.x + threadIdx.x;
  if (node >= n) return;
  const float* zr = z + (long)node * 64;
  float acc[OUT_DIM];
  #pragma unroll
  for (int j = 0; j < OUT_DIM; ++j) acc[j] = B[j];
  for (int k = 0; k < 64; ++k) {
    float zv = zr[k];
    #pragma unroll
    for (int j = 0; j < OUT_DIM; ++j) acc[j] += zv * W[k * OUT_DIM + j];
  }
  float* o = out + (long)node * OUT_DIM;
  #pragma unroll
  for (int j = 0; j < OUT_DIM; ++j) o[j] = acc[j];
}

// ---------------------------------------------------------------- host

extern "C" void kernel_launch(void* const* d_in, const int* in_sizes, int n_in,
                              void* d_out, int out_size, void* d_ws, size_t ws_size,
                              hipStream_t stream) {
  const float* x      = (const float*)d_in[0];
  const int*   eidx   = (const int*)  d_in[1];
  const float* ew     = (const float*)d_in[2];
  const float* sp_w1  = (const float*)d_in[3];
  const float* sp_b1  = (const float*)d_in[4];
  const float* sp_w2  = (const float*)d_in[5];
  const float* sp_b2  = (const float*)d_in[6];
  const float* fe_w   = (const float*)d_in[7];
  const float* fe_b   = (const float*)d_in[8];
  const float* c1_wl  = (const float*)d_in[9];
  const float* c1_wr  = (const float*)d_in[10];
  const float* c1_b   = (const float*)d_in[11];
  const float* c2_w   = (const float*)d_in[12];
  const float* c2_b   = (const float*)d_in[13];
  const float* c3_wl  = (const float*)d_in[14];
  const float* c3_wr  = (const float*)d_in[15];
  const float* c3_b   = (const float*)d_in[16];
  const float* c4_w   = (const float*)d_in[17];
  const float* c4_b   = (const float*)d_in[18];
  const float* bn_g   = (const float*)d_in[19];
  const float* bn_b   = (const float*)d_in[20];
  const float* fu_w   = (const float*)d_in[21];
  const float* fu_b   = (const float*)d_in[22];
  const float* cl_w1  = (const float*)d_in[23];
  const float* cl_b1  = (const float*)d_in[24];
  const float* cl_w2  = (const float*)d_in[25];
  const float* cl_b2  = (const float*)d_in[26];

  const int N = in_sizes[0] / 19;
  const int E = in_sizes[1] / 2;
  const int* src = eidx;
  const int* dst = eidx + E;

  // ---- workspace layout
  char* base = (char*)d_ws;
  size_t off = 0;
  auto alloc = [&](size_t bytes) -> void* {
    off = (off + 255) & ~(size_t)255;
    void* p = base + off;
    off += bytes;
    return p;
  };
  float* h0   = (float*)alloc((size_t)N * FUSED_DIM * 4);
  float* hA   = (float*)alloc((size_t)N * H_DIM * 4);   // h1, later h3
  float* h2   = (float*)alloc((size_t)N * H_DIM * 4);
  float* h4   = (float*)alloc((size_t)N * H_DIM * 4);
  float* acc  = (float*)alloc((size_t)N * H_DIM * 4);   // scatter accum, later z
  float* tmp  = (float*)alloc((size_t)N * H_DIM * 4);   // gemm outputs
  float* cnt  = (float*)alloc((size_t)N * 4);
  float* invc = (float*)alloc((size_t)N * 4);
  float* deg  = (float*)alloc((size_t)N * 4);
  float* dis  = (float*)alloc((size_t)N * 4);
  float* bns  = (float*)alloc(H_DIM * 4);
  float* bnq  = (float*)alloc(H_DIM * 4);
  _Float16* wc1l = (_Float16*)alloc((size_t)FUSED_DIM * H_DIM * 2);
  _Float16* wc1r = (_Float16*)alloc((size_t)FUSED_DIM * H_DIM * 2);
  _Float16* wc2  = (_Float16*)alloc((size_t)H_DIM * H_DIM * 2);
  _Float16* wc3l = (_Float16*)alloc((size_t)H_DIM * H_DIM * 2);
  _Float16* wc3r = (_Float16*)alloc((size_t)H_DIM * H_DIM * 2);
  _Float16* wc4  = (_Float16*)alloc((size_t)H_DIM * H_DIM * 2);
  _Float16* wfuA = (_Float16*)alloc((size_t)H_DIM * H_DIM * 2);
  _Float16* wfuB = (_Float16*)alloc((size_t)H_DIM * H_DIM * 2);
  _Float16* wcl1 = (_Float16*)alloc((size_t)H_DIM * 64 * 2);

  auto pack = [&](const float* s, _Float16* d, int K, int ncols) {
    int nel = K * ncols;
    pack_b_kernel<<<(nel + 255) / 256, 256, 0, stream>>>(s, d, K, ncols);
  };
  auto fill = [&](float* p, float v, long n) {
    fill_kernel<<<(int)((n + 255) / 256), 256, 0, stream>>>(p, v, n);
  };

  // ---- weight packing into WMMA fragment order (tiny, once per launch)
  pack(c1_wl, wc1l, FUSED_DIM, H_DIM);
  pack(c1_wr, wc1r, FUSED_DIM, H_DIM);
  pack(c2_w,  wc2,  H_DIM, H_DIM);
  pack(c3_wl, wc3l, H_DIM, H_DIM);
  pack(c3_wr, wc3r, H_DIM, H_DIM);
  pack(c4_w,  wc4,  H_DIM, H_DIM);
  pack(fu_w,                 wfuA, H_DIM, H_DIM);   // top half: multiplies h2
  pack(fu_w + H_DIM * H_DIM, wfuB, H_DIM, H_DIM);   // bottom half: multiplies h4
  pack(cl_w1, wcl1, H_DIM, 64);

  // ---- embedding
  embed_kernel<<<(N + 127) / 128, 128, 0, stream>>>(x, sp_w1, sp_b1, sp_w2, sp_b2, fe_w, fe_b, h0, N);

  // ---- degrees & normalization terms (graph-invariant across convs)
  fill(cnt, 0.f, N);
  fill(deg, 1.f, N);  // self-loop weight folded in
  degree_kernel<<<(E + 255) / 256, 256, 0, stream>>>(dst, ew, cnt, deg, E);
  node_prep_kernel<<<(N + 255) / 256, 256, 0, stream>>>(cnt, deg, invc, dis, N);

  const int gemmGrid = (N + 15) / 16;
  const int edgeGrid = (E + 255) / 256;
  const float invN = 1.f / (float)N;
  const long NH = (long)N * H_DIM;

  auto bn = [&](const float* in, const float* g, const float* b, float* outp) {
    fill(bns, 0.f, H_DIM);
    fill(bnq, 0.f, H_DIM);
    bn_stats_kernel<<<512, 256, 0, stream>>>(in, bns, bnq, N);
    bn_apply_relu_kernel<<<(int)((NH + 255) / 256), 256, 0, stream>>>(in, bns, bnq, g, b, outp, NH, invN);
  };

  // ---- conv1: SAGE on h0 (K=96) -> hA (h1)
  fill(acc, 0.f, (long)N * FUSED_DIM);
  scatter_sum_kernel<FUSED_DIM><<<edgeGrid, 256, 0, stream>>>(h0, src, dst, acc, E);
  wmma_gemm_kernel<FUSED_DIM, true, false><<<gemmGrid, 256, 0, stream>>>(
      acc, invc, (const v16h*)wc1l, h0, (const v16h*)wc1r, c1_b, tmp, N, H_DIM);
  bn(tmp, bn_g + 0 * H_DIM, bn_b + 0 * H_DIM, hA);

  // ---- conv2: GCN on h1 (K=128) -> h2
  wmma_gemm_kernel<H_DIM, false, false><<<gemmGrid, 256, 0, stream>>>(
      hA, nullptr, (const v16h*)wc2, nullptr, nullptr, nullptr, tmp, N, H_DIM);
  gcn_selfinit_kernel<<<(int)(((long)N * (H_DIM / 4) + 255) / 256), 256, 0, stream>>>(tmp, dis, c2_b, acc, N);
  gcn_scatter_kernel<<<edgeGrid, 256, 0, stream>>>(tmp, src, dst, ew, dis, acc, E);
  bn(acc, bn_g + 1 * H_DIM, bn_b + 1 * H_DIM, h2);

  // ---- conv3: SAGE on h2 (K=128) -> hA (h3)
  fill(acc, 0.f, NH);
  scatter_sum_kernel<H_DIM><<<edgeGrid, 256, 0, stream>>>(h2, src, dst, acc, E);
  wmma_gemm_kernel<H_DIM, true, false><<<gemmGrid, 256, 0, stream>>>(
      acc, invc, (const v16h*)wc3l, h2, (const v16h*)wc3r, c3_b, tmp, N, H_DIM);
  bn(tmp, bn_g + 2 * H_DIM, bn_b + 2 * H_DIM, hA);

  // ---- conv4: GCN on h3 (K=128) -> h4
  wmma_gemm_kernel<H_DIM, false, false><<<gemmGrid, 256, 0, stream>>>(
      hA, nullptr, (const v16h*)wc4, nullptr, nullptr, nullptr, tmp, N, H_DIM);
  gcn_selfinit_kernel<<<(int)(((long)N * (H_DIM / 4) + 255) / 256), 256, 0, stream>>>(tmp, dis, c4_b, acc, N);
  gcn_scatter_kernel<<<edgeGrid, 256, 0, stream>>>(tmp, src, dst, ew, dis, acc, E);
  bn(acc, bn_g + 3 * H_DIM, bn_b + 3 * H_DIM, h4);

  // ---- fusion: tmp = [h2|h4] @ fu_w + fu_b  (two-operand WMMA GEMM, K=128 each)
  wmma_gemm_kernel<H_DIM, true, false><<<gemmGrid, 256, 0, stream>>>(
      h2, nullptr, (const v16h*)wfuA, h4, (const v16h*)wfuB, fu_b, tmp, N, H_DIM);

  // ---- classifier stage 1: acc(z) = relu(tmp @ cl_w1 + cl_b1), 64 cols -> 4 waves / block
  wmma_gemm_kernel<H_DIM, false, true><<<gemmGrid, 128, 0, stream>>>(
      tmp, nullptr, (const v16h*)wcl1, nullptr, nullptr, cl_b1, acc, N, 64);

  // ---- classifier stage 2: out = z @ cl_w2 + cl_b2
  cls2_kernel<<<(N + 127) / 128, 128, 0, stream>>>(acc, cl_w2, cl_b2, (float*)d_out, N);
}
